// ThetaGammaCLEVRN_16286515986782
// MI455X (gfx1250) — matrix operand order repaired and checked by reading, not compiled, not run on hardware
//
#include <hip/hip_runtime.h>
#include <hip/hip_bf16.h>

typedef __attribute__((ext_vector_type(16))) _Float16 v16h;
typedef __attribute__((ext_vector_type(8)))  _Float16 v8h;
typedef __attribute__((ext_vector_type(8)))  float    v8f;

#define NT 16
#define NG 64
#define HID 64
#define NOBJ 10
#define QDIM 60
#define NSTEPS 20
#define DTSTEP 0.01f
#define KPACK 160              // 140 effective K, zero-padded to 5*32
#define BPK_ELEMS (5*4*32*16)  // chunks * ntiles * lanes * halves = 10240

// ---------------------------------------------------------------------------
// Prep: build Weff = [W1[0:80]; Wq @ W1[80:144]; 0-pad] in WMMA-B lane layout
// (fp16), and beff = b1 + bq @ W1[80:144] (fp32). ~20.7 KB of workspace.
// ---------------------------------------------------------------------------
__global__ void tg_prep(const float* __restrict__ Wq, const float* __restrict__ W1,
                        const float* __restrict__ b1, const float* __restrict__ bq,
                        _Float16* __restrict__ Bpk, float* __restrict__ beff) {
    int id = blockIdx.x * 256 + threadIdx.x;
    if (id < BPK_ELEMS) {
        int i    =  id        & 15;   // half index within lane
        int lane = (id >> 4)  & 31;
        int t    = (id >> 9)  & 3;    // N tile
        int c    =  id >> 11;         // K chunk
        int K = c * 32 + (i >> 3) * 16 + (lane >> 4) * 8 + (i & 7);   // B layout: lanes 0-15 K=0..7,16..23; 16-31 K=8..15,24..31
        int N = t * 16 + (lane & 15);
        float v = 0.0f;
        if (K < 80) {
            v = W1[K * HID + N];
        } else if (K < 140) {
            int d = K - 80;
            for (int j = 0; j < HID; ++j)
                v = fmaf(Wq[d * HID + j], W1[(80 + j) * HID + N], v);
        }
        Bpk[id] = (_Float16)v;
    }
    if (id < HID) {
        float v = b1[id];
        for (int j = 0; j < HID; ++j)
            v = fmaf(bq[j], W1[(80 + j) * HID + id], v);
        beff[id] = v;
    }
}

__device__ __forceinline__ float wave_sum(float v) {
#pragma unroll
    for (int m = 16; m >= 1; m >>= 1) v += __shfl_xor(v, m, 32);
    return v;
}

// ---------------------------------------------------------------------------
// Fused: oscillator integration (one wave32 per sample) -> LDS A tile ->
// WMMA f16 GEMM (K=160) -> ReLU -> 64->2 head -> log_softmax.
// 16 waves/block = one 16-sample WMMA M-tile per block.
// ---------------------------------------------------------------------------
__global__ __launch_bounds__(512, 1) void tg_main(
    const float* __restrict__ scene, const float* __restrict__ query,
    const float* __restrict__ tph0, const float* __restrict__ tfr,
    const float* __restrict__ gph0, const float* __restrict__ gfr,
    const float* __restrict__ Ws,   const float* __restrict__ bs,
    const _Float16* __restrict__ Bpk, const float* __restrict__ beff,
    const float* __restrict__ W2,   const float* __restrict__ b2,
    float* __restrict__ out) {
    __shared__ __align__(16) _Float16 As[16][KPACK];   // 5 KB: A tile (16 samples x 160)
    __shared__ float Hs[16][HID];                      // 4 KB: hidden activations

    const int wave = threadIdx.x >> 5;
    const int lane = threadIdx.x & 31;
    const int b    = blockIdx.x * 16 + wave;
    const bool isT = lane < NT;

    // ---------------- per-sample oscillator state ----------------
    float tp = 0.f, tf = 0.f, ta = 1.f;
    if (isT) { tp = tph0[b * NT + lane]; tf = tfr[b * NT + lane]; }
    const int g0 = lane, g1 = lane + 32;
    float gp0v = gph0[b * NG + g0], gp1v = gph0[b * NG + g1];
    float gf0  = gfr [b * NG + g0], gf1  = gfr [b * NG + g1];

    // scene mean over objects (dims live on lanes 0-15)
    float sm = 0.f;
    if (isT) {
#pragma unroll
        for (int o = 0; o < NOBJ; ++o) sm += scene[(b * NOBJ + o) * 16 + lane];
        sm *= (1.0f / NOBJ);
    }
    // gamma_amp0 = sigmoid(scene_mean @ Ws + bs)
    float d0 = bs[g0], d1 = bs[g1];
#pragma unroll
    for (int d = 0; d < 16; ++d) {
        float s = __shfl(sm, d, 32);
        d0 = fmaf(s, Ws[d * NG + g0], d0);
        d1 = fmaf(s, Ws[d * NG + g1], d1);
    }
    float ga0 = 1.f / (1.f + expf(-d0));
    float ga1 = 1.f / (1.f + expf(-d1));

    // ---------------- Euler integration, 20 steps ----------------
    for (int s = 0; s < NSTEPS; ++s) {
        float st = 0.f, ct = 0.f;
        if (isT) __sincosf(tp, &st, &ct);
        float sg0, cg0, sg1, cg1;
        __sincosf(gp0v, &sg0, &cg0);
        __sincosf(gp1v, &sg1, &cg1);

        float Ti = wave_sum(isT ? ta * st : 0.f);
        float Tr = wave_sum(isT ? ta * ct : 0.f);
        float mi = wave_sum(isT ? st : 0.f) * (1.0f / NT);
        float mr = wave_sum(isT ? ct : 0.f) * (1.0f / NT);
        float Gi = wave_sum(ga0 * sg0 + ga1 * sg1);
        float Gr = wave_sum(ga0 * cg0 + ga1 * cg1);

        if (isT) {
            float coup_t = (Ti * ct - Tr * st) * (1.0f / NT);   // K_THETA = 1
            tp += DTSTEP * (tf + coup_t);
            ta += DTSTEP * (ta * (1.f - ta * ta));
        }
        float c0 = (Gi * cg0 - Gr * sg0) * (1.0f / NG);         // K_GAMMA = 1
        float x0 = mi * cg0 - mr * sg0;                          // K_CROSS = 0.5
        gp0v += DTSTEP * (gf0 + c0 + 0.5f * x0);
        ga0  += DTSTEP * (ga0 * (1.f - ga0 * ga0));
        float c1 = (Gi * cg1 - Gr * sg1) * (1.0f / NG);
        float x1 = mi * cg1 - mr * sg1;
        gp1v += DTSTEP * (gf1 + c1 + 0.5f * x1);
        ga1  += DTSTEP * (ga1 * (1.f - ga1 * ga1));
    }

    // ---------------- stage A row: [ta | ga | query | 0] -> fp16 LDS ----------------
    if (isT) As[wave][lane] = (_Float16)ta;
    As[wave][NT + g0] = (_Float16)ga0;
    As[wave][NT + g1] = (_Float16)ga1;
    As[wave][80 + lane] = (_Float16)query[b * QDIM + lane];
    if (lane + 32 < QDIM) As[wave][80 + lane + 32] = (_Float16)query[b * QDIM + lane + 32];
    if (lane < 20) As[wave][140 + lane] = (_Float16)0.0f;
    __syncthreads();

    // ---------------- WMMA GEMM: (16 x 160) x (160 x 64), waves 0-3 ----------------
    if (wave < 4) {
        const int t   = wave;        // N tile
        const int row = lane & 15;   // A layout: M = lane % 16
        const int hi  = lane >> 4;   // K half-select
        v8f acc = {};
#pragma unroll
        for (int c = 0; c < 5; ++c) {
            v8h lo8 = *(const v8h*)&As[row][c * 32 + hi * 8];        // K = c*32 + {0..7 | 8..15}
            v8h hi8 = *(const v8h*)&As[row][c * 32 + 16 + hi * 8];   // K = c*32 + {16..23 | 24..31}
            v16h a  = __builtin_shufflevector(lo8, hi8,
                        0, 1, 2, 3, 4, 5, 6, 7, 8, 9, 10, 11, 12, 13, 14, 15);
            v16h bm = *(const v16h*)&Bpk[((c * 4 + t) * 32 + lane) * 16];
            acc = __builtin_amdgcn_wmma_f32_16x16x32_f16(
                    false, a, false, bm, (short)0, acc, false, false);
        }
        // C layout: VGPR r, lanes 0-15: M=r, N=lane; lanes 16-31: M=8+r, N=lane-16
        float bias = beff[t * 16 + (lane & 15)];
#pragma unroll
        for (int r = 0; r < 8; ++r) {
            float h = acc[r] + bias;
            Hs[r + 8 * hi][t * 16 + (lane & 15)] = fmaxf(h, 0.0f);
        }
    }
    __syncthreads();

    // ---------------- head: h @ W2 + b2, log_softmax ----------------
    if (wave == 0 && lane < 16) {
        const int m  = lane;
        const int bo = blockIdx.x * 16 + m;
        float l0 = b2[0], l1 = b2[1];
#pragma unroll 8
        for (int n = 0; n < HID; ++n) {
            float h = Hs[m][n];
            l0 = fmaf(h, W2[n * 2 + 0], l0);
            l1 = fmaf(h, W2[n * 2 + 1], l1);
        }
        float mx  = fmaxf(l0, l1);
        float lse = mx + logf(expf(l0 - mx) + expf(l1 - mx));
        out[bo * 2 + 0] = l0 - lse;
        out[bo * 2 + 1] = l1 - lse;
    }
}

extern "C" void kernel_launch(void* const* d_in, const int* in_sizes, int n_in,
                              void* d_out, int out_size, void* d_ws, size_t ws_size,
                              hipStream_t stream) {
    const float* scene = (const float*)d_in[0];
    const float* query = (const float*)d_in[1];
    const float* tph0  = (const float*)d_in[2];
    const float* tfr   = (const float*)d_in[3];
    const float* gph0  = (const float*)d_in[4];
    const float* gfr   = (const float*)d_in[5];
    const float* Ws    = (const float*)d_in[6];
    const float* bs    = (const float*)d_in[7];
    const float* Wq    = (const float*)d_in[8];
    const float* bq    = (const float*)d_in[9];
    const float* W1    = (const float*)d_in[10];
    const float* b1    = (const float*)d_in[11];
    const float* W2    = (const float*)d_in[12];
    const float* b2    = (const float*)d_in[13];
    float* out = (float*)d_out;

    _Float16* Bpk = (_Float16*)d_ws;
    float*    beff = (float*)((char*)d_ws + BPK_ELEMS * sizeof(_Float16));

    const int B = out_size / 2;           // 65536
    tg_prep<<<(BPK_ELEMS + 255) / 256, 256, 0, stream>>>(Wq, W1, b1, bq, Bpk, beff);
    tg_main<<<B / 16, 512, 0, stream>>>(scene, query, tph0, tfr, gph0, gfr,
                                        Ws, bs, Bpk, beff, W2, b2, out);
}
